// DrugGRU_71674414236061
// MI455X (gfx1250) — matrix operand order repaired and checked by reading, not compiled, not run on hardware
//
#include <hip/hip_runtime.h>
#include <hip/hip_bf16.h>

// ---------------------------------------------------------------------------
// GRU forward (embedding -> input GEMM via WMMA bf16 -> sequential scan with
// LDS-resident w_hh slices across 16 cooperating workgroups -> BCE loss).
// Target: MI455X / gfx1250 (wave32, WMMA, 320KB LDS per WGP).
// ---------------------------------------------------------------------------

constexpr int kSeq = 4096;
constexpr int kH   = 512;
constexpr int kG3  = 3 * kH;      // 1536
constexpr int kWgs = 16;          // workgroups cooperating on the scan

typedef __attribute__((ext_vector_type(16))) __bf16    v16bf;
typedef __attribute__((ext_vector_type(8)))  float     v8f;
typedef __attribute__((ext_vector_type(8)))  unsigned  v8u;

// Pack two fp32 into one packed-bf16 dword (truncate) with a single
// v_perm_b32: dst = { hi16(f1), hi16(f0) }.
__device__ __forceinline__ unsigned pack2bf(float f0, float f1) {
  return __builtin_amdgcn_perm(__builtin_bit_cast(unsigned, f1),   // S0
                               __builtin_bit_cast(unsigned, f0),   // S1
                               0x07060302u);
}

// ---------------------------------------------------------------------------
// Kernel 1: gx[s, 0:1536] = emb[x[s]] @ w_ih.T + b_ih
// Grid: (1536/128, 4096/32). Block: 256 threads = 8 waves, 2(M) x 4(N).
// Each wave owns a 16x32 output tile (two 16x16 WMMA accumulators) so one
// A fragment (embedding gather) feeds two v_wmma_f32_16x16x32_bf16.
// ---------------------------------------------------------------------------
__global__ void __launch_bounds__(256)
gru_gx_wmma_kernel(const long long* __restrict__ x,
                   const float* __restrict__ emb,
                   const float* __restrict__ w_ih,
                   const float* __restrict__ b_ih,
                   float* __restrict__ gx)
{
  const int lane  = threadIdx.x & 31;
  const int wave  = threadIdx.x >> 5;
  const int mBase = blockIdx.y * 32 + (wave >> 2) * 16;
  const int nBase = blockIdx.x * 128 + (wave & 3) * 32;   // two 16-wide tiles

  const int grp  = lane >> 4;           // 0: lanes 0-15, 1: lanes 16-31
  const int mRow = mBase + (lane & 15); // A row this lane carries
  const int nRc  = lane & 15;           // column-in-tile this lane carries

  const float* __restrict__ arow  = emb  + (size_t)x[mRow] * kH;        // gather
  const float* __restrict__ brow0 = w_ih + (size_t)(nBase + nRc) * kH;  // tile 0
  const float* __restrict__ brow1 = w_ih + (size_t)(nBase + 16 + nRc) * kH;

  v8f c0 = {};
  v8f c1 = {};
  #pragma unroll 2
  for (int k0 = 0; k0 < kH; k0 += 32) {
    // A 16x32 bf16: lane grp 0 -> K k0+0..15 ; grp 1 -> K k0+16..31
    const int ka = k0 + (grp << 4);
    // B 32x16 bf16: lane grp 0 -> K k0+0..15 ; grp 1 -> K k0+16..31
    const int kb = k0 + (grp << 4);

    const float4 a0 = *(const float4*)(arow + ka);
    const float4 a1 = *(const float4*)(arow + ka + 4);
    const float4 a2 = *(const float4*)(arow + ka + 8);
    const float4 a3 = *(const float4*)(arow + ka + 12);
    const float4 p0 = *(const float4*)(brow0 + kb);
    const float4 p1 = *(const float4*)(brow0 + kb + 4);
    const float4 p2 = *(const float4*)(brow0 + kb + 8);
    const float4 p3 = *(const float4*)(brow0 + kb + 12);
    const float4 q0 = *(const float4*)(brow1 + kb);
    const float4 q1 = *(const float4*)(brow1 + kb + 4);
    const float4 q2 = *(const float4*)(brow1 + kb + 8);
    const float4 q3 = *(const float4*)(brow1 + kb + 12);

    if (k0 + 32 < kH) {                       // gfx1250 global_prefetch_b8
      __builtin_prefetch(arow + ka + 32, 0, 1);
      __builtin_prefetch(brow0 + kb + 32, 0, 1);
      __builtin_prefetch(brow1 + kb + 32, 0, 1);
    }

    v8u Au, B0u, B1u;
    Au[0] = pack2bf(a0.x, a0.y);  Au[1] = pack2bf(a0.z, a0.w);
    Au[2] = pack2bf(a1.x, a1.y);  Au[3] = pack2bf(a1.z, a1.w);
    Au[4] = pack2bf(a2.x, a2.y);  Au[5] = pack2bf(a2.z, a2.w);
    Au[6] = pack2bf(a3.x, a3.y);  Au[7] = pack2bf(a3.z, a3.w);
    B0u[0] = pack2bf(p0.x, p0.y); B0u[1] = pack2bf(p0.z, p0.w);
    B0u[2] = pack2bf(p1.x, p1.y); B0u[3] = pack2bf(p1.z, p1.w);
    B0u[4] = pack2bf(p2.x, p2.y); B0u[5] = pack2bf(p2.z, p2.w);
    B0u[6] = pack2bf(p3.x, p3.y); B0u[7] = pack2bf(p3.z, p3.w);
    B1u[0] = pack2bf(q0.x, q0.y); B1u[1] = pack2bf(q0.z, q0.w);
    B1u[2] = pack2bf(q1.x, q1.y); B1u[3] = pack2bf(q1.z, q1.w);
    B1u[4] = pack2bf(q2.x, q2.y); B1u[5] = pack2bf(q2.z, q2.w);
    B1u[6] = pack2bf(q3.x, q3.y); B1u[7] = pack2bf(q3.z, q3.w);

    const v16bf A  = __builtin_bit_cast(v16bf, Au);
    const v16bf B0 = __builtin_bit_cast(v16bf, B0u);
    const v16bf B1 = __builtin_bit_cast(v16bf, B1u);

    c0 = __builtin_amdgcn_wmma_f32_16x16x32_bf16(
        false, A, false, B0, (short)0, c0, false, false);
    c1 = __builtin_amdgcn_wmma_f32_16x16x32_bf16(
        false, A, false, B1, (short)0, c1, false, false);
  }

  // C/D layout: VGPR v, lane L -> M = v + 8*(L>=16), N = L&15.
  const int n0   = nBase + nRc;
  const int n1   = nBase + 16 + nRc;
  const float b0 = b_ih[n0];
  const float b1 = b_ih[n1];
  const int mOff = mBase + (grp << 3);
  #pragma unroll
  for (int v = 0; v < 8; ++v) {
    gx[(size_t)(mOff + v) * kG3 + n0] = c0[v] + b0;
    gx[(size_t)(mOff + v) * kG3 + n1] = c1[v] + b1;
  }
}

// ---------------------------------------------------------------------------
// Kernel 2: sequential GRU scan. 16 workgroups x 256 threads.
// WG w owns hidden units j in [32w, 32w+32) and keeps its 96 w_hh rows
// (r/z/n for those units: 96*512 fp32 = 192KB) resident in LDS (320KB WGP).
// Per step: 8 threads/unit split K; LDS reduce; gate math; 2KB h exchanged
// through L2 under a monotonic device-wide barrier. WG0 fuses decoder + BCE.
// ---------------------------------------------------------------------------
__global__ void __launch_bounds__(256)
gru_scan_kernel(const float* __restrict__ gx,
                const float* __restrict__ w_hh,
                const float* __restrict__ b_hh,
                const float* __restrict__ dec_w,
                const float* __restrict__ dec_b,
                const float* __restrict__ target,
                float* __restrict__ gH,         // [512] shared hidden state
                unsigned* __restrict__ counter, // monotonic barrier counter
                float* __restrict__ out)
{
  extern __shared__ float smem[];
  float* sW = smem;                 // 96*512 floats (192KB) w_hh slice
  float* sh = sW + 96 * kH;         // 512 floats, current hidden state
  float* sb = sh + kH;              // 96 floats, b_hh slice
  float* pp = sb + 96;              // 768 floats, partial sums / reduce

  const int tid = threadIdx.x;
  const int wg  = blockIdx.x;       // 0..15
  const int u   = tid >> 3;         // 0..31  unit within this WG
  const int s   = tid & 7;          // 0..7   K-split slot
  const int j   = wg * 32 + u;      // global hidden unit index

  // Stage w_hh slice into LDS: local row lr = set*32 + uu  <->  global row
  // set*512 + wg*32 + uu  (set 0=r, 1=z, 2=n).
  for (int idx = tid; idx < 96 * kH; idx += 256) {
    const int lr  = idx >> 9;
    const int k   = idx & (kH - 1);
    const int set = lr >> 5;
    const int uu  = lr & 31;
    sW[idx] = w_hh[(size_t)(set * kH + wg * 32 + uu) * kH + k];
  }
  if (tid < 96) {
    const int set = tid >> 5, uu = tid & 31;
    sb[tid] = b_hh[set * kH + wg * 32 + uu];
  }
  for (int k = tid; k < kH; k += 256) sh[k] = 0.0f;
  __syncthreads();

  const float* __restrict__ wr = sW + (size_t)(0 * 32 + u) * kH;
  const float* __restrict__ wz = sW + (size_t)(1 * 32 + u) * kH;
  const float* __restrict__ wn = sW + (size_t)(2 * 32 + u) * kH;
  const int k0 = s * 64;

  for (int t = 0; t < kSeq; ++t) {
    // partial dot products over this thread's 64-wide K slice
    float ar = 0.f, az = 0.f, an = 0.f;
    #pragma unroll 8
    for (int k = 0; k < 64; ++k) {
      const float hv = sh[k0 + k];
      ar = fmaf(wr[k0 + k], hv, ar);
      az = fmaf(wz[k0 + k], hv, az);
      an = fmaf(wn[k0 + k], hv, an);
    }
    pp[tid]       = ar;
    pp[256 + tid] = az;
    pp[512 + tid] = an;
    __syncthreads();

    if (s == 0) {
      float gr = 0.f, gz = 0.f, gn = 0.f;
      #pragma unroll
      for (int q = 0; q < 8; ++q) {
        gr += pp[u * 8 + q];
        gz += pp[256 + u * 8 + q];
        gn += pp[512 + u * 8 + q];
      }
      gr += sb[u];
      gz += sb[32 + u];
      gn += sb[64 + u];
      const float* __restrict__ gxt = gx + (size_t)t * kG3;
      const float r    = 1.0f / (1.0f + __expf(-(gxt[j] + gr)));
      const float z    = 1.0f / (1.0f + __expf(-(gxt[kH + j] + gz)));
      const float n    = tanhf(gxt[2 * kH + j] + r * gn);
      const float hnew = (1.0f - z) * n + z * sh[j];
      gH[j] = hnew;   // publish this WG's 32 units
    }
    __syncthreads();

    // device-wide barrier: monotonic counter, acquire-load spin + s_sleep
    if (tid == 0) {
      __threadfence();
      atomicAdd(counter, 1u);
      const unsigned goal = (unsigned)kWgs * (unsigned)(t + 1);
      while (__hip_atomic_load(counter, __ATOMIC_ACQUIRE,
                               __HIP_MEMORY_SCOPE_AGENT) < goal) {
        __builtin_amdgcn_s_sleep(1);
      }
      __threadfence();
    }
    __syncthreads();

    for (int k = tid; k < kH; k += 256) sh[k] = gH[k];  // pull full h (2KB, L2)
    __syncthreads();
  }

  // decoder logit + stable BCE, fused on WG0
  if (wg == 0) {
    float acc = 0.f;
    for (int k = tid; k < kH; k += 256) acc = fmaf(sh[k], dec_w[k], acc);
    pp[tid] = acc;
    __syncthreads();
    for (int off = 128; off > 0; off >>= 1) {
      if (tid < off) pp[tid] += pp[tid + off];
      __syncthreads();
    }
    if (tid == 0) {
      const float logit = pp[0] + dec_b[0];
      const float tgt   = target[0];
      auto logsig = [](float v) {
        return fminf(v, 0.0f) - log1pf(__expf(-fabsf(v)));
      };
      out[0] = -(tgt * logsig(logit) + (1.0f - tgt) * logsig(-logit));
    }
  }
}

// ---------------------------------------------------------------------------
extern "C" void kernel_launch(void* const* d_in, const int* in_sizes, int n_in,
                              void* d_out, int out_size, void* d_ws, size_t ws_size,
                              hipStream_t stream) {
  (void)in_sizes; (void)n_in; (void)out_size; (void)ws_size;

  const long long* x    = (const long long*)d_in[0]; // int64 token ids
  const float* target   = (const float*)d_in[1];
  const float* emb      = (const float*)d_in[2];
  const float* w_ih     = (const float*)d_in[3];
  const float* w_hh     = (const float*)d_in[4];
  const float* b_ih     = (const float*)d_in[5];
  const float* b_hh     = (const float*)d_in[6];
  const float* dec_w    = (const float*)d_in[7];
  const float* dec_b    = (const float*)d_in[8];
  float* out            = (float*)d_out;

  // workspace layout: gx (24MB) | gH (2KB) | barrier counter
  char* ws = (char*)d_ws;
  float* gx       = (float*)ws;
  float* gH       = (float*)(ws + (size_t)kSeq * kG3 * sizeof(float));
  unsigned* cnt   = (unsigned*)(ws + (size_t)kSeq * kG3 * sizeof(float) + 2048);

  hipMemsetAsync(cnt, 0, sizeof(unsigned), stream);

  dim3 gemmGrid(kG3 / 128, kSeq / 32);   // 12 x 128 blocks
  gru_gx_wmma_kernel<<<gemmGrid, 256, 0, stream>>>(x, emb, w_ih, b_ih, gx);

  const size_t shmem = (size_t)(96 * kH + kH + 96 + 768) * sizeof(float); // ~198KB
  gru_scan_kernel<<<kWgs, 256, shmem, stream>>>(gx, w_hh, b_hh, dec_w, dec_b,
                                                target, gH, cnt, out);
}
